// MatrixGAT_9801115369778
// MI455X (gfx1250) — matrix operand order repaired and checked by reading, not compiled, not run on hardware
//
#include <hip/hip_runtime.h>

// ---------------------------------------------------------------------------
// MatrixGAT on MI455X (gfx1250): conv folded into GEMM1, bf16 WMMA GEMMs,
// atomic segment-softmax edge phase. Round 2: branch-free GEMM inner loop,
// float4-vectorized edge gathers.
// ---------------------------------------------------------------------------

typedef __attribute__((ext_vector_type(16))) __bf16 v16bf;
typedef __attribute__((ext_vector_type(8)))  float  v8f;

#define NN      50000
#define EE      800000
#define ETOT    (EE + NN)          // with self loops
#define K1      1536               // folded conv: K drops 4096 -> 1536
#define NO1     512                // xl1 | xr1 concatenated
#define HC1     256                // 4 heads * 64
#define K2      256
#define NO2     256                // xl2 | xr2 concatenated
#define HC2     128                // 1 head * 128

// ------------------------- float atomic max --------------------------------
__device__ inline void atomicMaxF32(float* addr, float value) {
    if (value >= 0.f)
        atomicMax((int*)addr, __float_as_int(value));
    else
        atomicMin((unsigned int*)addr, __float_as_uint(value));
}

__device__ inline void edge_pair(int e, const int* __restrict__ ei, int& s, int& d) {
    if (e < EE) { s = ei[e]; d = ei[EE + e]; }
    else        { s = e - EE; d = s; }        // self loop
}

// ------------------------- weight preparation ------------------------------
// Packed B-fragment layout (bf16 16x16x32 WMMA, wave32):
//   fragment (kb, nb): 32 lanes x 16 bf16, lane l -> column n = l&15,
//   K group kg = l>>4; dword d holds K = kb*32 + kg*16 + 2d (+1 in hi half).
//   Stored contiguously: elem index = ((kb*NB + nb)*32 + lane)*16 + 2d + dk.

// GEMM1 weights: Weff2[c*24+t][o] = sum_co conv_w[co][k] * W[co*64+p][o],
// where j = t*64+c, p = j/24, k = j%24, W = [Wl1 | Wr1].
__global__ void prep_w1(const float* __restrict__ Wl, const float* __restrict__ Wr,
                        const float* __restrict__ convw, __bf16* __restrict__ Wp) {
    int q = blockIdx.x * blockDim.x + threadIdx.x;        // one dword pair
    const int NB = NO1 / 16;
    if (q >= (K1 / 32) * NB * 32 * 8) return;
    int d = q & 7, rest = q >> 3;
    int lane = rest & 31; rest >>= 5;
    int nb = rest % NB, kb = rest / NB;
    int col = nb * 16 + (lane & 15);
    const float* Wsrc = (col < 256) ? Wl : Wr;
    int ccol = col & 255;
#pragma unroll
    for (int dk = 0; dk < 2; ++dk) {
        int kk = kb * 32 + (lane >> 4) * 16 + 2 * d + dk; // row in [0,1536)
        int c = kk / 24, t = kk - 24 * c;
        int j = t * 64 + c;
        int p = j / 24, k = j - 24 * p;
        float acc = 0.f;
        for (int co = 0; co < 64; ++co)
            acc += convw[co * 24 + k] * Wsrc[(co * 64 + p) * 256 + ccol];
        Wp[(size_t)q * 2 + dk] = (__bf16)acc;
    }
}

// GEMM2 weights: plain [Wl2 | Wr2] -> packed bf16 fragments.
__global__ void prep_w2(const float* __restrict__ Wl, const float* __restrict__ Wr,
                        __bf16* __restrict__ Wp) {
    int q = blockIdx.x * blockDim.x + threadIdx.x;
    const int NB = NO2 / 16;
    if (q >= (K2 / 32) * NB * 32 * 8) return;
    int d = q & 7, rest = q >> 3;
    int lane = rest & 31; rest >>= 5;
    int nb = rest % NB, kb = rest / NB;
    int col = nb * 16 + (lane & 15);
    const float* Wsrc = (col < 128) ? Wl : Wr;
    int ccol = col & 127;
#pragma unroll
    for (int dk = 0; dk < 2; ++dk) {
        int kk = kb * 32 + (lane >> 4) * 16 + 2 * d + dk;
        Wp[(size_t)q * 2 + dk] = (__bf16)Wsrc[kk * 128 + ccol];
    }
}

// Folded bias for GEMM1: beff[o] = bl/br[o] + sum_m conv_b[m>>6] * W[m][o]
__global__ void prep_beff(const float* __restrict__ bl, const float* __restrict__ br,
                          const float* __restrict__ convb,
                          const float* __restrict__ Wl, const float* __restrict__ Wr,
                          float* __restrict__ beff) {
    int o = blockIdx.x * blockDim.x + threadIdx.x;
    if (o >= NO1) return;
    const float* Wsrc = (o < 256) ? Wl : Wr;
    int col = o & 255;
    float acc = (o < 256) ? bl[o] : br[o - 256];
    for (int m = 0; m < 4096; ++m) acc += convb[m >> 6] * Wsrc[m * 256 + col];
    beff[o] = acc;
}

__global__ void prep_b2(const float* __restrict__ bl, const float* __restrict__ br,
                        float* __restrict__ b) {
    int o = blockIdx.x * blockDim.x + threadIdx.x;
    if (o >= NO2) return;
    b[o] = (o < 128) ? bl[o] : br[o - 128];
}

// ------------------------------ WMMA GEMM ----------------------------------
// Block = 256 threads = 8 waves; computes a 16-row x NOUT panel.
// A panel (16 x K, f32) is staged once into LDS as bf16 (padded stride),
// each wave owns NT 16-column tiles; per k-step: 2 ds_load_b128 A-frag +
// NT x 2 global b128 B-frags + NT v_wmma_f32_16x16x32_bf16.
// Inner loop is branch-free (clamped prefetch index) so the NT loop fully
// unrolls and software-pipelines without accumulator copies.
template <int NT>
__global__ void gemm_wmma(const float* __restrict__ A, int K,
                          const __bf16* __restrict__ Bp,
                          const float* __restrict__ bias,
                          float* __restrict__ C, int NOUT) {
    extern __shared__ __bf16 As[];                 // [16][K+8] (+8 pad: no bank conflicts)
    const int strideA = K + 8;
    const int m0 = blockIdx.x * 16;
    const int tid = threadIdx.x;

    // Stage + convert A panel: f32 global -> bf16 LDS (A read from HBM once).
    const int total = 16 * K;
    for (int idx = tid * 4; idx < total; idx += 256 * 4) {
        int row = idx / K;
        int col = idx - row * K;
        const float4 f = *(const float4*)(A + (size_t)(m0 + row) * K + col);
        __bf16* dst = &As[row * strideA + col];
        dst[0] = (__bf16)f.x; dst[1] = (__bf16)f.y;
        dst[2] = (__bf16)f.z; dst[3] = (__bf16)f.w;
    }
    __syncthreads();

    const int lane = tid & 31;
    const int wave = tid >> 5;
    const int mrow = lane & 15;
    const int kg   = lane >> 4;
    const int NB   = NOUT >> 4;

    // Per-lane invariant parts of the addresses.
    const __bf16* __restrict__ apBase = &As[mrow * strideA + kg * 16];
    const __bf16* __restrict__ bpLane = Bp + (size_t)lane * 16;

    v8f acc[NT];
#pragma unroll
    for (int t = 0; t < NT; ++t) acc[t] = {};

    const int ksteps = K >> 5;
    for (int kb = 0; kb < ksteps; ++kb) {
        // Branch-free prefetch of next k-step's B fragments (clamped index).
        const int kbn = (kb + 1 < ksteps) ? (kb + 1) : kb;
#pragma unroll
        for (int t = 0; t < NT; ++t)
            __builtin_prefetch(bpLane + (size_t)(kbn * NB + wave * NT + t) * 512, 0, 3);

        // A fragment (contiguous K order, matching packed B): lane holds row mrow,
        // K = kb*32 + kg*16 + [0..15] in v0..7.
        union { uint4 u[2]; v16bf v; } af;
        const __bf16* ap = apBase + kb * 32;
        af.u[0] = *(const uint4*)(ap);
        af.u[1] = *(const uint4*)(ap + 8);

#pragma unroll
        for (int t = 0; t < NT; ++t) {
            const __bf16* bp = bpLane + (size_t)(kb * NB + wave * NT + t) * 512;
            union { uint4 u[2]; v16bf v; } bf;
            bf.u[0] = *(const uint4*)(bp);
            bf.u[1] = *(const uint4*)(bp + 8);
            acc[t] = __builtin_amdgcn_wmma_f32_16x16x32_bf16(
                false, af.v, false, bf.v, (short)0, acc[t], false, false);
        }
    }

    // C/D 16x16 f32 layout: lane&15 = column, kg*8+r = row (VGPR r).
#pragma unroll
    for (int t = 0; t < NT; ++t) {
        const int nb  = wave * NT + t;
        const int col = nb * 16 + (lane & 15);
        const float b = bias[col];
        const int mbase = m0 + kg * 8;
#pragma unroll
        for (int r = 0; r < 8; ++r)
            C[(size_t)(mbase + r) * NOUT + col] = acc[t][r] + b;
    }
}

// ------------------------------ edge phase ---------------------------------
__global__ void fill_f32(float* __restrict__ p, float v, int n) {
    int i = blockIdx.x * blockDim.x + threadIdx.x;
    if (i < n) p[i] = v;
}

// score = att . leakyrelu(xl[src] + xr[dst]); running segment max over dst.
// float4 gathers: rows are 16B-aligned (C % 4 == 0).
__global__ void edge_scores(const float* __restrict__ XL, const int* __restrict__ ei,
                            const float* __restrict__ att,
                            float* __restrict__ score, float* __restrict__ smax,
                            int H, int C) {
    int idx = blockIdx.x * blockDim.x + threadIdx.x;
    if (idx >= ETOT * H) return;
    int e = idx / H, h = idx - e * H;
    int s, d; edge_pair(e, ei, s, d);
    const int HC = H * C, stride = 2 * HC;
    const float4* xl4 = (const float4*)(XL + (size_t)s * stride + h * C);
    const float4* xr4 = (const float4*)(XL + (size_t)d * stride + HC + h * C);
    const float4* a4  = (const float4*)(att + h * C);
    float acc = 0.f;
    for (int c = 0; c < (C >> 2); ++c) {
        float4 l = xl4[c], r = xr4[c], a = a4[c];
        float v0 = l.x + r.x; v0 = (v0 > 0.f) ? v0 : 0.2f * v0;
        float v1 = l.y + r.y; v1 = (v1 > 0.f) ? v1 : 0.2f * v1;
        float v2 = l.z + r.z; v2 = (v2 > 0.f) ? v2 : 0.2f * v2;
        float v3 = l.w + r.w; v3 = (v3 > 0.f) ? v3 : 0.2f * v3;
        acc += a.x * v0 + a.y * v1 + a.z * v2 + a.w * v3;
    }
    score[idx] = acc;
    atomicMaxF32(&smax[d * H + h], acc);
}

// ex = exp(score - smax[dst]); denom[dst] += ex   (score overwritten with ex)
__global__ void edge_exp(const int* __restrict__ ei, float* __restrict__ score,
                         const float* __restrict__ smax, float* __restrict__ denom,
                         int H) {
    int idx = blockIdx.x * blockDim.x + threadIdx.x;
    if (idx >= ETOT * H) return;
    int e = idx / H, h = idx - e * H;
    int s, d; edge_pair(e, ei, s, d);
    (void)s;
    float ex = __expf(score[idx] - smax[d * H + h]);
    score[idx] = ex;
    atomicAdd(&denom[d * H + h], ex);
}

// out[dst] += (ex/denom[dst]) * xl[src]; 4 channels per thread (float4 gather).
__global__ void edge_scatter(const float* __restrict__ XL, const int* __restrict__ ei,
                             const float* __restrict__ exs, const float* __restrict__ denom,
                             float* __restrict__ out, int H, int C) {
    int idx = blockIdx.x * blockDim.x + threadIdx.x;   // Etot * HC/4 threads
    const int HC = H * C;
    const int W4 = HC >> 2;
    int e = idx / W4;
    if (e >= ETOT) return;
    int i4 = (idx - e * W4) << 2;                      // channel base (multiple of 4)
    int h = i4 / C;
    int s, d; edge_pair(e, ei, s, d);
    float alpha = exs[e * H + h] / denom[d * H + h];
    float4 v = *(const float4*)(XL + (size_t)s * 2 * HC + i4);
    float* o = out + (size_t)d * HC + i4;
    atomicAdd(o + 0, alpha * v.x);
    atomicAdd(o + 1, alpha * v.y);
    atomicAdd(o + 2, alpha * v.z);
    atomicAdd(o + 3, alpha * v.w);
}

__global__ void relu_bias_inplace(float* __restrict__ out, const float* __restrict__ bias,
                                  int total, int HC) {
    int idx = blockIdx.x * blockDim.x + threadIdx.x;
    if (idx >= total) return;
    float v = out[idx] + bias[idx % HC];
    out[idx] = v > 0.f ? v : 0.f;
}

__global__ void add_bias_inplace(float* __restrict__ out, const float* __restrict__ bias,
                                 int total, int HC) {
    int idx = blockIdx.x * blockDim.x + threadIdx.x;
    if (idx >= total) return;
    out[idx] += bias[idx % HC];
}

// ------------------------------ launch -------------------------------------
extern "C" void kernel_launch(void* const* d_in, const int* in_sizes, int n_in,
                              void* d_out, int out_size, void* d_ws, size_t ws_size,
                              hipStream_t stream) {
    const float* x     = (const float*)d_in[0];   // [N,1536] row-major == A of GEMM1
    const int*   ei    = (const int*)  d_in[1];   // [2,E]
    const float* convw = (const float*)d_in[2];
    const float* convb = (const float*)d_in[3];
    const float* Wl1   = (const float*)d_in[4];
    const float* bl1   = (const float*)d_in[5];
    const float* Wr1   = (const float*)d_in[6];
    const float* br1   = (const float*)d_in[7];
    const float* att1  = (const float*)d_in[8];
    const float* bias1 = (const float*)d_in[9];
    const float* Wl2   = (const float*)d_in[10];
    const float* bl2   = (const float*)d_in[11];
    const float* Wr2   = (const float*)d_in[12];
    const float* br2   = (const float*)d_in[13];
    const float* att2  = (const float*)d_in[14];
    const float* bias2 = (const float*)d_in[15];
    float* out = (float*)d_out;                   // [N,128]

    // workspace carve-up (256B aligned)
    char* ws = (char*)d_ws;
    size_t cur = 0;
    auto alloc = [&](size_t bytes) {
        char* p = ws + cur;
        cur += (bytes + 255) & ~size_t(255);
        return p;
    };
    __bf16* Wp1   = (__bf16*)alloc((size_t)K1 * NO1 * 2);
    __bf16* Wp2   = (__bf16*)alloc((size_t)K2 * NO2 * 2);
    float*  beff  = (float*) alloc(NO1 * 4);
    float*  bcat2 = (float*) alloc(NO2 * 4);
    float*  XL1   = (float*) alloc((size_t)NN * NO1 * 4);
    float*  out1  = (float*) alloc((size_t)NN * HC1 * 4);
    float*  XL2   = (float*) alloc((size_t)NN * NO2 * 4);
    float*  sc1   = (float*) alloc((size_t)ETOT * 4 * 4);
    float*  sc2   = (float*) alloc((size_t)ETOT * 4);
    float*  smax1 = (float*) alloc((size_t)NN * 4 * 4);
    float*  den1  = (float*) alloc((size_t)NN * 4 * 4);
    float*  smax2 = (float*) alloc((size_t)NN * 4);
    float*  den2  = (float*) alloc((size_t)NN * 4);
    (void)ws_size; (void)in_sizes; (void)n_in; (void)out_size;

    const int T = 256;
    auto blocks = [](long long n, int t) { return (unsigned)((n + t - 1) / t); };

    // ---- weight prep (conv folded into projection) ----
    prep_w1  <<<blocks((K1/32)*(NO1/16)*32*8, T), T, 0, stream>>>(Wl1, Wr1, convw, Wp1);
    prep_w2  <<<blocks((K2/32)*(NO2/16)*32*8, T), T, 0, stream>>>(Wl2, Wr2, Wp2);
    prep_beff<<<blocks(NO1, T), T, 0, stream>>>(bl1, br1, convb, Wl1, Wr1, beff);
    prep_b2  <<<blocks(NO2, T), T, 0, stream>>>(bl2, br2, bcat2);

    // ---- GEMM1: [N,1536] x [1536,512] -> xl1|xr1 ----
    gemm_wmma<4><<<NN / 16, T, 16 * (K1 + 8) * 2, stream>>>(x, K1, Wp1, beff, XL1, NO1);

    // ---- GATv2 layer 1 edge phase ----
    fill_f32<<<blocks(NN * 4, T), T, 0, stream>>>(smax1, -1e30f, NN * 4);
    fill_f32<<<blocks(NN * 4, T), T, 0, stream>>>(den1, 0.f, NN * 4);
    fill_f32<<<blocks((long long)NN * HC1, T), T, 0, stream>>>(out1, 0.f, NN * HC1);
    edge_scores <<<blocks((long long)ETOT * 4, T), T, 0, stream>>>(XL1, ei, att1, sc1, smax1, 4, 64);
    edge_exp    <<<blocks((long long)ETOT * 4, T), T, 0, stream>>>(ei, sc1, smax1, den1, 4);
    edge_scatter<<<blocks((long long)ETOT * (HC1 / 4), T), T, 0, stream>>>(XL1, ei, sc1, den1, out1, 4, 64);
    relu_bias_inplace<<<blocks((long long)NN * HC1, T), T, 0, stream>>>(out1, bias1, NN * HC1, HC1);

    // ---- GEMM2: [N,256] x [256,256] -> xl2|xr2 ----
    gemm_wmma<2><<<NN / 16, T, 16 * (K2 + 8) * 2, stream>>>(out1, K2, Wp2, bcat2, XL2, NO2);

    // ---- GATv2 layer 2 edge phase (accumulate into d_out) ----
    fill_f32<<<blocks(NN, T), T, 0, stream>>>(smax2, -1e30f, NN);
    fill_f32<<<blocks(NN, T), T, 0, stream>>>(den2, 0.f, NN);
    fill_f32<<<blocks((long long)NN * HC2, T), T, 0, stream>>>(out, 0.f, NN * HC2);
    edge_scores <<<blocks((long long)ETOT, T), T, 0, stream>>>(XL2, ei, att2, sc2, smax2, 1, 128);
    edge_exp    <<<blocks((long long)ETOT, T), T, 0, stream>>>(ei, sc2, smax2, den2, 1);
    edge_scatter<<<blocks((long long)ETOT * (HC2 / 4), T), T, 0, stream>>>(XL2, ei, sc2, den2, out, 1, 128);
    add_bias_inplace<<<blocks((long long)NN * HC2, T), T, 0, stream>>>(out, bias2, NN * HC2, HC2);
}
